// ALiBiMultiHeadAttention_26929444946438
// MI455X (gfx1250) — matrix operand order repaired and checked by reading, not compiled, not run on hardware
//
#include <hip/hip_runtime.h>

// ---------------------------------------------------------------------------
// ALiBi causal multi-head attention, MI455X (gfx1250, wave32, WMMA bf16)
//   x      : [4096, 1024] fp32
//   w_kqv  : [1024, 3072] fp32   (columns: [K | Q | V], each 1024 = 16 heads x 64)
//   out    : [4096, 1024] fp32
// Pipeline: cvt x->bf16 ; transpose w->bf16 ; WMMA QKV GEMM (writes Q,K [h][s][d]
// and V^T [h][d][s] in bf16) ; flash-attention with v_wmma_f32_16x16x32_bf16.
// ---------------------------------------------------------------------------

#define SEQ    4096
#define DMODEL 1024
#define NHEADS 16
#define DHEAD  64
#define N3     3072

typedef __attribute__((ext_vector_type(16))) __bf16 bf16x16;
typedef __attribute__((ext_vector_type(8)))  __bf16 bf16x8;
typedef __attribute__((ext_vector_type(4)))  __bf16 bf16x4;
typedef __attribute__((ext_vector_type(8)))  float  f32x8;

__device__ __forceinline__ f32x8 wmma_bf16(bf16x16 a, bf16x16 b, f32x8 c) {
  // D = A(16x32 bf16) * B(32x16 bf16) + C(16x16 f32)
  return __builtin_amdgcn_wmma_f32_16x16x32_bf16(
      /*neg_a=*/false, a, /*neg_b=*/false, b,
      /*c_mod=*/(short)0, c, /*reuse_a=*/false, /*reuse_b=*/false);
}

__device__ __forceinline__ bf16x16 combine8(bf16x8 lo, bf16x8 hi) {
  bf16x16 r;
#pragma unroll
  for (int i = 0; i < 8; ++i) { r[i] = lo[i]; r[i + 8] = hi[i]; }
  return r;
}

// --------------------------- x (fp32) -> xb (bf16) -------------------------
__global__ void __launch_bounds__(256) cvt_x_kernel(const float* __restrict__ x,
                                                    __bf16* __restrict__ xb) {
  size_t i = ((size_t)blockIdx.x * blockDim.x + threadIdx.x) * 4;
  float4 v = *(const float4*)(x + i);
  bf16x4 o;
  o[0] = (__bf16)v.x; o[1] = (__bf16)v.y;
  o[2] = (__bf16)v.z; o[3] = (__bf16)v.w;
  *(bf16x4*)(xb + i) = o;
}

// ----------------- w [1024][3072] fp32 -> wt [3072][1024] bf16 -------------
__global__ void __launch_bounds__(256) trw_kernel(const float* __restrict__ w,
                                                  __bf16* __restrict__ wt) {
  __shared__ __bf16 t[32][33];
  const int nb = blockIdx.x * 32;   // N (3072)
  const int kb = blockIdx.y * 32;   // K (1024)
  const int tx = threadIdx.x & 31;
  const int ty = threadIdx.x >> 5;  // 0..7
#pragma unroll
  for (int j = ty; j < 32; j += 8)
    t[j][tx] = (__bf16)w[(size_t)(kb + j) * N3 + nb + tx];
  __syncthreads();
#pragma unroll
  for (int j = ty; j < 32; j += 8)
    wt[(size_t)(nb + j) * DMODEL + kb + tx] = t[tx][j];
}

// --------------------------- QKV projection GEMM ---------------------------
// One wave computes a 16(M) x 64(N) tile, K-loop step 32, WMMA bf16.
// Output columns [0,1024)->K, [1024,2048)->Q (stored [h][s][dh]),
// [2048,3072)->V stored transposed [h][dh][s].
__global__ void __launch_bounds__(128) qkv_gemm_kernel(
    const __bf16* __restrict__ xb, const __bf16* __restrict__ wt,
    __bf16* __restrict__ Kb, __bf16* __restrict__ Qb, __bf16* __restrict__ Vt) {
  __shared__ __bf16 ldsC[4][16 * 64];
  const int lane = threadIdx.x & 31;
  const int wave = threadIdx.x >> 5;
  const int ln   = lane & 15;
  const int hi   = lane >> 4;
  const int m0   = blockIdx.x * 16;
  const int n0   = (blockIdx.y * 4 + wave) * 64;

  const __bf16* xrow = xb + (size_t)(m0 + ln) * DMODEL;
  const __bf16* wcol[4];
#pragma unroll
  for (int nt = 0; nt < 4; ++nt)
    wcol[nt] = wt + (size_t)(n0 + nt * 16 + ln) * DMODEL + hi * 16;

  f32x8 acc[4] = {};
  for (int k = 0; k < DMODEL; k += 32) {
    // A fragment 16x32: lanes 0-15 row=ln K 0-15; lanes 16-31 K 16-31 (contiguous)
    bf16x8 alo = *(const bf16x8*)(xrow + k + hi * 16);
    bf16x8 ahi = *(const bf16x8*)(xrow + k + hi * 16 + 8);
    bf16x16 a = combine8(alo, ahi);
#pragma unroll
    for (int nt = 0; nt < 4; ++nt) {
      // B fragment 32x16: lane col=ln, K contiguous (lo half K0-15, hi half K16-31)
      bf16x16 b = *(const bf16x16*)(wcol[nt] + k);
      acc[nt] = wmma_bf16(a, b, acc[nt]);
    }
  }

  const int matrix = n0 / DMODEL;            // 0=K, 1=Q, 2=V
  const int h      = (n0 % DMODEL) / DHEAD;  // one head per 64-col wave tile

  if (matrix == 2) {
    // V^T[h][d][s]: fp32 C-layout gives 8 consecutive rows per lane -> pack 16B store
#pragma unroll
    for (int nt = 0; nt < 4; ++nt) {
      const int d = nt * 16 + ln;
      bf16x8 pk;
#pragma unroll
      for (int r = 0; r < 8; ++r) pk[r] = (__bf16)acc[nt][r];
      *(bf16x8*)(Vt + ((size_t)h * DHEAD + d) * SEQ + m0 + 8 * hi) = pk;
    }
  } else {
    // Q/K [h][s][dh]: scatter C-layout into wave-private LDS, gather row-major
    __bf16* dst = (matrix == 1) ? Qb : Kb;
#pragma unroll
    for (int nt = 0; nt < 4; ++nt)
#pragma unroll
      for (int r = 0; r < 8; ++r)
        ldsC[wave][(r + 8 * hi) * 64 + nt * 16 + ln] = (__bf16)acc[nt][r];
    asm volatile("s_wait_dscnt 0x0" ::: "memory");  // DS in-order within wave
    const __bf16* src = &ldsC[wave][ln * 64 + hi * 32];
    bf16x16 c0 = *(const bf16x16*)(src);
    bf16x16 c1 = *(const bf16x16*)(src + 16);
    __bf16* gdst = dst + ((size_t)h * SEQ + m0 + ln) * DHEAD + hi * 32;
    *(bf16x16*)(gdst) = c0;
    *(bf16x16*)(gdst + 16) = c1;
  }
}

// ------------------------- flash attention (per wave: 16 queries) ----------
__global__ void __launch_bounds__(256) attn_kernel(
    const __bf16* __restrict__ Qb, const __bf16* __restrict__ Kb,
    const __bf16* __restrict__ Vt, float* __restrict__ out) {
  __shared__ __bf16 ldsP[8][16 * 32];   // per-wave P-tile transpose buffer
  const int lane = threadIdx.x & 31;
  const int wave = threadIdx.x >> 5;
  const int ln   = lane & 15;
  const int hi   = lane >> 4;
  const int h    = blockIdx.y;
  const int q0   = (blockIdx.x * 8 + wave) * 16;

  // slopes m_h = (2^8)^(1/16) ^ -(h+1) = 2^(-(h+1)/2); score scaled by 1/sqrt(1024)
  const float slope     = exp2f(-0.5f * (float)(h + 1));
  const float inv_scale = 1.0f / 32.0f;

  // Q A-fragments for d in [0,32) and [32,64)
  const __bf16* qrow = Qb + ((size_t)h * SEQ + q0 + ln) * DHEAD;
  bf16x16 qa0 = combine8(*(const bf16x8*)(qrow + hi * 16),
                         *(const bf16x8*)(qrow + hi * 16 + 8));
  bf16x16 qa1 = combine8(*(const bf16x8*)(qrow + 32 + hi * 16),
                         *(const bf16x8*)(qrow + 40 + hi * 16));

  f32x8 o[4] = {};
  float m[8], l[8];
#pragma unroll
  for (int r = 0; r < 8; ++r) { m[r] = -1e30f; l[r] = 0.0f; }

  const __bf16* kbase = Kb + (size_t)h * SEQ * DHEAD;
  const __bf16* vbase = Vt + (size_t)h * DHEAD * SEQ;

  for (int kb = 0; kb < q0 + 16; kb += 32) {
    // ---- S = Q K^T for keys [kb, kb+32), two 16-key N-tiles ----
    f32x8 s[2];
#pragma unroll
    for (int t = 0; t < 2; ++t) {
      const __bf16* krow = kbase + (size_t)(kb + t * 16 + ln) * DHEAD + hi * 16;
      bf16x16 b0 = *(const bf16x16*)(krow);        // d 0..31
      bf16x16 b1 = *(const bf16x16*)(krow + 32);   // d 32..63
      f32x8 z = {};
      s[t] = wmma_bf16(qa0, b0, z);
      s[t] = wmma_bf16(qa1, b1, s[t]);
    }
    if (kb + 32 < q0 + 16)
      __builtin_prefetch(kbase + (size_t)(kb + 32 + ln) * DHEAD, 0, 1);

    // ---- online softmax: bias, causal mask, row max/sum over 16-lane groups ----
#pragma unroll
    for (int r = 0; r < 8; ++r) {
      const int i  = q0 + r + 8 * hi;
      const int j0 = kb + ln, j1 = kb + 16 + ln;
      float v0 = s[0][r] * inv_scale + slope * (float)(j0 - i);
      float v1 = s[1][r] * inv_scale + slope * (float)(j1 - i);
      if (j0 > i) v0 = -1e30f;
      if (j1 > i) v1 = -1e30f;
      float mx = fmaxf(v0, v1);
      mx = fmaxf(mx, __shfl_xor(mx, 1, 32));
      mx = fmaxf(mx, __shfl_xor(mx, 2, 32));
      mx = fmaxf(mx, __shfl_xor(mx, 4, 32));
      mx = fmaxf(mx, __shfl_xor(mx, 8, 32));
      const float nm = fmaxf(m[r], mx);
      const float al = __expf(m[r] - nm);
      m[r] = nm;
      const float p0 = __expf(v0 - nm);
      const float p1 = __expf(v1 - nm);
      float rs = p0 + p1;
      rs += __shfl_xor(rs, 1, 32);
      rs += __shfl_xor(rs, 2, 32);
      rs += __shfl_xor(rs, 4, 32);
      rs += __shfl_xor(rs, 8, 32);
      l[r] = l[r] * al + rs;
#pragma unroll
      for (int nt = 0; nt < 4; ++nt) o[nt][r] *= al;
      // P tile (C-layout) -> LDS row-major [16][32] in bf16
      ldsP[wave][(r + 8 * hi) * 32 + ln]      = (__bf16)p0;
      ldsP[wave][(r + 8 * hi) * 32 + 16 + ln] = (__bf16)p1;
    }
    asm volatile("s_wait_dscnt 0x0" ::: "memory");  // DS in-order within wave

    // P as A-fragment (K-dim = 32 keys of this block)
    const __bf16* prow = &ldsP[wave][ln * 32];
    bf16x16 pa = combine8(*(const bf16x8*)(prow + hi * 16),
                          *(const bf16x8*)(prow + hi * 16 + 8));

    // ---- O += P V  (B-fragments contiguous from V^T[h][d][s]) ----
#pragma unroll
    for (int nt = 0; nt < 4; ++nt) {
      const __bf16* vrow = vbase + (size_t)(nt * 16 + ln) * SEQ + kb + hi * 16;
      bf16x16 vb = *(const bf16x16*)(vrow);
      o[nt] = wmma_bf16(pa, vb, o[nt]);
    }
  }

  // ---- epilogue: divide by softmax denominator, write fp32 output ----
#pragma unroll
  for (int r = 0; r < 8; ++r) {
    const float rl = 1.0f / l[r];
    const int i = q0 + r + 8 * hi;
#pragma unroll
    for (int nt = 0; nt < 4; ++nt)
      out[(size_t)i * DMODEL + h * DHEAD + nt * 16 + ln] = o[nt][r] * rl;
  }
}

// ---------------------------------------------------------------------------
extern "C" void kernel_launch(void* const* d_in, const int* in_sizes, int n_in,
                              void* d_out, int out_size, void* d_ws, size_t ws_size,
                              hipStream_t stream) {
  (void)in_sizes; (void)n_in; (void)out_size; (void)ws_size;
  const float* x = (const float*)d_in[0];   // [4096,1024]
  const float* w = (const float*)d_in[1];   // [1024,3072]
  float* out = (float*)d_out;               // [4096,1024]

  // workspace layout (bf16): xb 8MB | wt 6MB | Kb 8MB | Qb 8MB | Vt 8MB
  __bf16* xb = (__bf16*)d_ws;
  __bf16* wt = xb + (size_t)SEQ * DMODEL;
  __bf16* Kb = wt + (size_t)N3 * DMODEL;
  __bf16* Qb = Kb + (size_t)SEQ * DMODEL;
  __bf16* Vt = Qb + (size_t)SEQ * DMODEL;

  cvt_x_kernel<<<(SEQ * DMODEL) / (256 * 4), 256, 0, stream>>>(x, xb);
  trw_kernel<<<dim3(N3 / 32, DMODEL / 32), 256, 0, stream>>>(w, wt);
  qkv_gemm_kernel<<<dim3(SEQ / 16, N3 / 256), 128, 0, stream>>>(xb, wt, Kb, Qb, Vt);
  attn_kernel<<<dim3(SEQ / 128, NHEADS), 256, 0, stream>>>(Qb, Kb, Vt, out);
}
